// DeepseekV2Attention_72138270703843
// MI455X (gfx1250) — matrix operand order repaired and checked by reading, not compiled, not run on hardware
//
#include <hip/hip_runtime.h>

// ---------------------------------------------------------------------------
// DeepSeek-V2 MLA attention for gfx1250 (MI455X), bf16 WMMA pipeline with
// async global->LDS staging (ASYNCcnt) for non-transposed tiles.
// ---------------------------------------------------------------------------

typedef __bf16 bf16_t;
typedef __attribute__((ext_vector_type(16))) __bf16 v16bf;
typedef __attribute__((ext_vector_type(8)))  float  v8f;

#define T_SEQ 2048
#define D_MODEL 5120
#define H_HEADS 32
#define NOPE 128
#define ROPE_D 64
#define QK_D 192          // NOPE + ROPE
#define KVR 512
#define QR 1536
#define VH 128
#define HQK (H_HEADS * QK_D)        // 6144
#define KVL (KVR + ROPE_D)          // 576
#define KVN (H_HEADS * (NOPE + VH)) // 8192
#define HV (H_HEADS * VH)           // 4096

// ---------------------------------------------------------------------------
// Async global->LDS copy (per-lane 16B), tracked by ASYNCcnt.
// ---------------------------------------------------------------------------
__device__ __forceinline__ void async_load_b128(unsigned lds_off, const void* gptr) {
    unsigned long long ga = (unsigned long long)(size_t)gptr;
    asm volatile("global_load_async_to_lds_b128 %0, %1, off"
                 :: "v"(lds_off), "v"(ga)
                 : "memory");
}
__device__ __forceinline__ void wait_asynccnt0() {
    asm volatile("s_wait_asynccnt 0" ::: "memory");
}

// ---------------------------------------------------------------------------
// WMMA fragment helpers (wave32).  A and B both use contiguous k = 16*half + s order.
// ---------------------------------------------------------------------------
__device__ __forceinline__ v16bf load_a_frag(const bf16_t* __restrict__ s, int ld, int lane) {
    int half = lane >> 4, r = lane & 15;
    const bf16_t* p = s + r * ld;
    v16bf a;
#pragma unroll
    for (int v = 0; v < 8; ++v) {
        int k = 2 * v + 16 * half;
        a[2 * v]     = p[k];
        a[2 * v + 1] = p[k + 1];
    }
    return a;
}

// B fragment from a K-contiguous (transposed) tile: element [k][c] at s[c*ld+k].
__device__ __forceinline__ v16bf load_bT_frag(const bf16_t* __restrict__ s, int ld, int lane) {
    int half = lane >> 4, c = lane & 15;
    const bf16_t* p = s + c * ld + 16 * half;
    v16bf b;
#pragma unroll
    for (int v = 0; v < 8; ++v) {
        b[2 * v]     = p[2 * v];
        b[2 * v + 1] = p[2 * v + 1];
    }
    return b;
}

__device__ __forceinline__ v8f wmma_bf16(v16bf a, v16bf b, v8f c) {
    return __builtin_amdgcn_wmma_f32_16x16x32_bf16(false, a, false, b, (short)0, c,
                                                   false, false);
}

// ---------------------------------------------------------------------------
// f32 -> bf16 cast
// ---------------------------------------------------------------------------
__global__ void cast_f32_bf16(const float* __restrict__ x, bf16_t* __restrict__ y, int n) {
    int i = (blockIdx.x * blockDim.x + threadIdx.x) * 4;
    if (i + 3 < n) {
        float4 v = *(const float4*)(x + i);
        y[i]     = (bf16_t)v.x;
        y[i + 1] = (bf16_t)v.y;
        y[i + 2] = (bf16_t)v.z;
        y[i + 3] = (bf16_t)v.w;
    } else {
        for (int j = i; j < n; ++j) y[j] = (bf16_t)x[j];
    }
}

// ---------------------------------------------------------------------------
// Generic bf16 GEMM: C(f32, MxN) = A(bf16, MxK) * B(bf16, KxN) [+ bias(N)]
// 128x128 block tile, K-step 32, 128 threads = 4 waves (2x2 wave grid),
// each wave computes 64x64 via 4x4 wmma tiles.  Double-buffered LDS.
// A tile staged with async global->LDS b128; B staged transposed via
// register 4x4 micro-transpose (async copies cannot transpose).
// NOTE: assumes M % 128 == 0 (true for all call sites: M = 2048).
// ---------------------------------------------------------------------------
#define GTM 128
#define GTN 128
#define GTK 32
#define LDA_S 40
#define LDB_T 40

__global__ __launch_bounds__(128) void gemm_bf16(
    const bf16_t* __restrict__ A, const bf16_t* __restrict__ B,
    const float* __restrict__ bias, float* __restrict__ C,
    int M, int N, int K) {
    __shared__ __align__(16) bf16_t As[2][GTM * LDA_S];
    __shared__ __align__(16) bf16_t BsT[2][GTN * LDB_T];

    int tid = threadIdx.x;
    int lane = tid & 31, wave = tid >> 5;        // 4 waves
    int wm = wave & 1, wn = wave >> 1;           // 2x2 wave grid
    int bm = blockIdx.y * GTM, bn = blockIdx.x * GTN;
    int half = lane >> 4, ln = lane & 15;

    v8f acc[4][4];
    v8f zero8 = {0.f, 0.f, 0.f, 0.f, 0.f, 0.f, 0.f, 0.f};
#pragma unroll
    for (int i = 0; i < 4; ++i)
#pragma unroll
        for (int j = 0; j < 4; ++j) acc[i][j] = zero8;

    ushort4 breg[8];

    // A tile: 128 rows x 32 cols = 8 KB = 512 x 16B chunks; 4 waves x 4 instrs.
    auto async_load_a = [&](int buf, int k0) {
        unsigned lds_base = (unsigned)(size_t)&As[buf][0];
#pragma unroll
        for (int p = 0; p < 4; ++p) {
            int q = (wave * 4 + p) * 32 + lane;   // chunk id 0..511
            int row = q >> 2;                     // 4 chunks (64B) per row
            int co = (q & 3) * 8;                 // element offset in row
            unsigned lds = lds_base + (unsigned)(row * LDA_S + co) * 2u;
            async_load_b128(lds, A + (size_t)(bm + row) * K + k0 + co);
        }
    };
    auto load_b_g = [&](int k0) {
#pragma unroll
        for (int m = 0; m < 2; ++m) {
            int mt = tid + m * 128;
            int kg = mt >> 5;           // 0..7  (k group of 4)
            int c0 = (mt & 31) * 4;     // 0..124
            int gc = bn + c0;
#pragma unroll
            for (int i = 0; i < 4; ++i) {
                ushort4 v = {0, 0, 0, 0};
                if (gc + 3 < N)
                    v = *(const ushort4*)(B + (size_t)(k0 + kg * 4 + i) * N + gc);
                breg[m * 4 + i] = v;
            }
        }
    };
    auto store_b_s = [&](int buf) {
#pragma unroll
        for (int m = 0; m < 2; ++m) {
            int mt = tid + m * 128;
            int kg = mt >> 5;
            int c0 = (mt & 31) * 4;
#pragma unroll
            for (int i = 0; i < 4; ++i) {   // i = column within micro-tile
                ushort4 o;
                o.x = ((const unsigned short*)&breg[m * 4 + 0])[i];
                o.y = ((const unsigned short*)&breg[m * 4 + 1])[i];
                o.z = ((const unsigned short*)&breg[m * 4 + 2])[i];
                o.w = ((const unsigned short*)&breg[m * 4 + 3])[i];
                *(ushort4*)(&BsT[buf][(c0 + i) * LDB_T + kg * 4]) = o;
            }
        }
    };

    int nk = K / GTK;
    async_load_a(0, 0);
    load_b_g(0);
    store_b_s(0);
    wait_asynccnt0();

    for (int kt = 0; kt < nk; ++kt) {
        __syncthreads();
        int cur = kt & 1;
        if (kt + 1 < nk) {
            async_load_a(cur ^ 1, (kt + 1) * GTK);
            load_b_g((kt + 1) * GTK);
        }
        if (kt + 2 < nk) {
            // prefetch B tile two steps ahead into GL2 (global_prefetch_b8)
            __builtin_prefetch(B + (size_t)((kt + 2) * GTK + (tid >> 5) * 4) * N + bn + (tid & 31) * 4, 0, 1);
        }

        v16bf af[4], bfr[4];
#pragma unroll
        for (int i = 0; i < 4; ++i)
            af[i] = load_a_frag(&As[cur][(wm * 64 + i * 16) * LDA_S], LDA_S, lane);
#pragma unroll
        for (int j = 0; j < 4; ++j)
            bfr[j] = load_bT_frag(&BsT[cur][(wn * 64 + j * 16) * LDB_T], LDB_T, lane);
#pragma unroll
        for (int i = 0; i < 4; ++i)
#pragma unroll
            for (int j = 0; j < 4; ++j)
                acc[i][j] = wmma_bf16(af[i], bfr[j], acc[i][j]);

        if (kt + 1 < nk) {
            store_b_s(cur ^ 1);
            wait_asynccnt0();   // own async A writes landed before publishing barrier
        }
    }

#pragma unroll
    for (int i = 0; i < 4; ++i) {
#pragma unroll
        for (int j = 0; j < 4; ++j) {
            int rbase = bm + wm * 64 + i * 16 + 8 * half;
            int col = bn + wn * 64 + j * 16 + ln;
            if (col < N) {
                float bv = bias ? bias[col] : 0.0f;
#pragma unroll
                for (int e = 0; e < 8; ++e) {
                    int row = rbase + e;
                    if (row < M) C[(size_t)row * N + col] = acc[i][j][e] + bv;
                }
            }
        }
    }
}

// ---------------------------------------------------------------------------
// RMSNorm: y(bf16) = x * rsqrt(mean(x^2)+eps) * g ; x has row stride xstride
// ---------------------------------------------------------------------------
__global__ __launch_bounds__(256) void rmsnorm_to_bf16(
    const float* __restrict__ x, int xstride, const float* __restrict__ g,
    bf16_t* __restrict__ y, int ncols) {
    int row = blockIdx.x, tid = threadIdx.x;
    const float* xr = x + (size_t)row * xstride;
    float ss = 0.f;
    for (int c = tid; c < ncols; c += 256) { float v = xr[c]; ss += v * v; }
    __shared__ float red[256];
    red[tid] = ss;
    __syncthreads();
    for (int s = 128; s > 0; s >>= 1) {
        if (tid < s) red[tid] += red[tid + s];
        __syncthreads();
    }
    float r = rsqrtf(red[0] / (float)ncols + 1e-6f);
    for (int c = tid; c < ncols; c += 256)
        y[(size_t)row * ncols + c] = (bf16_t)(xr[c] * r * g[c]);
}

// ---------------------------------------------------------------------------
// Assemble Q/K/V (bf16) with interleaved RoPE on q_pe and shared k_pe.
// Q,K: [T][H][192], V: [T][H][128]. One block per (t,h), 160 threads.
// ---------------------------------------------------------------------------
__global__ __launch_bounds__(160) void assemble_qkv(
    const float* __restrict__ q, const float* __restrict__ kv,
    const float* __restrict__ lat, const int* __restrict__ pos_ids,
    bf16_t* __restrict__ Q, bf16_t* __restrict__ K, bf16_t* __restrict__ V) {
    int t = blockIdx.x, h = blockIdx.y, tid = threadIdx.x;
    const float* qrow = q + (size_t)t * HQK + h * QK_D;
    const float* kvrow = kv + (size_t)t * KVN + h * (NOPE + VH);
    bf16_t* Qr = Q + ((size_t)t * H_HEADS + h) * QK_D;
    bf16_t* Kr = K + ((size_t)t * H_HEADS + h) * QK_D;
    bf16_t* Vr = V + ((size_t)t * H_HEADS + h) * VH;
    if (tid < 128) {
        Qr[tid] = (bf16_t)qrow[tid];
        Kr[tid] = (bf16_t)kvrow[tid];
        Vr[tid] = (bf16_t)kvrow[NOPE + tid];
    } else if (tid < 160) {
        int j = tid - 128; // 0..31 rope pair index
        float p = (float)pos_ids[t];
        float inv = __expf(-((float)j / 32.0f) * __logf(10000.0f));
        float f = p * inv;
        float c = cosf(f), s = sinf(f);
        float q1 = qrow[NOPE + 2 * j], q2 = qrow[NOPE + 2 * j + 1];
        Qr[NOPE + 2 * j]     = (bf16_t)(q1 * c - q2 * s);
        Qr[NOPE + 2 * j + 1] = (bf16_t)(q1 * s + q2 * c);
        const float* lrow = lat + (size_t)t * KVL + KVR;
        float k1 = lrow[2 * j], k2 = lrow[2 * j + 1];
        Kr[NOPE + 2 * j]     = (bf16_t)(k1 * c - k2 * s);
        Kr[NOPE + 2 * j + 1] = (bf16_t)(k1 * s + k2 * c);
    }
}

// ---------------------------------------------------------------------------
// Causal flash attention.  One block = 64 q-rows x one head, 4 waves; each
// wave owns 16 q rows.  K-blocks of 32.  S = Q·K^T and O += P·V via WMMA.
// Q and K tiles staged with async global->LDS; V staged transposed.
// ---------------------------------------------------------------------------
#define FA_QB 64
#define FA_KB 32
#define LDQS 200
#define LDVT 40
#define LDPS 40

__global__ __launch_bounds__(128) void flash_attn(
    const bf16_t* __restrict__ Q, const bf16_t* __restrict__ K,
    const bf16_t* __restrict__ V, bf16_t* __restrict__ O) {
    __shared__ __align__(16) bf16_t Qs[FA_QB * LDQS];
    __shared__ __align__(16) bf16_t Ks[FA_KB * LDQS];
    __shared__ __align__(16) bf16_t VsT[VH * LDVT];     // [feature][key]
    __shared__ __align__(16) bf16_t Ps[4 * 16 * LDPS];

    int tid = threadIdx.x, lane = tid & 31, wave = tid >> 5;
    int half = lane >> 4, ln = lane & 15;
    int qb0 = blockIdx.x * FA_QB;
    int h = blockIdx.y;

    // Async-load Q tile 64x192 (24 KB = 1536 chunks; 4 waves x 12 instrs)
    {
        unsigned lds_base = (unsigned)(size_t)&Qs[0];
#pragma unroll
        for (int p = 0; p < 12; ++p) {
            int q = (wave * 12 + p) * 32 + lane;   // 0..1535
            int row = q / 24;                      // 24 chunks (384B) per row
            int co = (q % 24) * 8;
            unsigned lds = lds_base + (unsigned)(row * LDQS + co) * 2u;
            async_load_b128(lds, Q + ((size_t)(qb0 + row) * H_HEADS + h) * QK_D + co);
        }
    }
    wait_asynccnt0();

    v8f oacc[8];
    v8f zero8 = {0.f, 0.f, 0.f, 0.f, 0.f, 0.f, 0.f, 0.f};
#pragma unroll
    for (int nt = 0; nt < 8; ++nt) oacc[nt] = zero8;
    float m_r[8], l_r[8];
#pragma unroll
    for (int e = 0; e < 8; ++e) { m_r[e] = -1e30f; l_r[e] = 0.f; }

    const float scale = 0.07216878364870323f; // 1/sqrt(192)
    int nkb = qb0 / FA_KB + 2;

    for (int kb = 0; kb < nkb; ++kb) {
        __syncthreads();
        // Async-load K block 32x192 (12 KB = 768 chunks; 4 waves x 6 instrs)
        {
            unsigned lds_base = (unsigned)(size_t)&Ks[0];
#pragma unroll
            for (int p = 0; p < 6; ++p) {
                int q = (wave * 6 + p) * 32 + lane;  // 0..767
                int row = q / 24;
                int co = (q % 24) * 8;
                unsigned lds = lds_base + (unsigned)(row * LDQS + co) * 2u;
                async_load_b128(lds, K + ((size_t)(kb * FA_KB + row) * H_HEADS + h) * QK_D + co);
            }
        }
        // Load V block 32x128 transposed via register 4x4 micro-transpose
#pragma unroll
        for (int m = 0; m < 2; ++m) {
            int mt = tid + m * 128;
            int kg = mt >> 5;           // key group 0..7
            int f0 = (mt & 31) * 4;     // feature 0..124
            ushort4 vr[4];
#pragma unroll
            for (int i = 0; i < 4; ++i)
                vr[i] = *(const ushort4*)(V + ((size_t)(kb * FA_KB + kg * 4 + i) * H_HEADS + h) * VH + f0);
#pragma unroll
            for (int i = 0; i < 4; ++i) {
                ushort4 o;
                o.x = ((const unsigned short*)&vr[0])[i];
                o.y = ((const unsigned short*)&vr[1])[i];
                o.z = ((const unsigned short*)&vr[2])[i];
                o.w = ((const unsigned short*)&vr[3])[i];
                *(ushort4*)(VsT + (f0 + i) * LDVT + kg * 4) = o;
            }
        }
        wait_asynccnt0();   // own K chunks landed before publishing barrier
        __syncthreads();

        // S tiles: 16 q rows x 32 k cols, contraction over d=192 in 6 chunks
        v8f s0 = zero8, s1 = zero8;
        const bf16_t* qsw = Qs + wave * 16 * LDQS;
#pragma unroll
        for (int d0 = 0; d0 < QK_D; d0 += 32) {
            v16bf a  = load_a_frag(qsw + d0, LDQS, lane);
            v16bf b0 = load_bT_frag(Ks + d0, LDQS, lane);
            v16bf b1 = load_bT_frag(Ks + 16 * LDQS + d0, LDQS, lane);
            s0 = wmma_bf16(a, b0, s0);
            s1 = wmma_bf16(a, b1, s1);
        }

        // Online softmax (rows live across 16-lane halves)
        int rowg0 = qb0 + wave * 16 + 8 * half;
        int colg0 = kb * FA_KB;
        bool need_mask = (colg0 + FA_KB - 1) > (qb0 + wave * 16);
#pragma unroll
        for (int e = 0; e < 8; ++e) {
            float v0 = s0[e] * scale, v1 = s1[e] * scale;
            if (need_mask) {
                int rg = rowg0 + e;
                if (colg0 + ln > rg) v0 = -1e30f;
                if (colg0 + 16 + ln > rg) v1 = -1e30f;
            }
            float rm = fmaxf(v0, v1);
            rm = fmaxf(rm, __shfl_xor(rm, 1, 16));
            rm = fmaxf(rm, __shfl_xor(rm, 2, 16));
            rm = fmaxf(rm, __shfl_xor(rm, 4, 16));
            rm = fmaxf(rm, __shfl_xor(rm, 8, 16));
            float mnew = fmaxf(m_r[e], rm);
            float alpha = __expf(m_r[e] - mnew);
            float e0 = __expf(v0 - mnew);
            float e1 = __expf(v1 - mnew);
            float rs = e0 + e1;
            rs += __shfl_xor(rs, 1, 16);
            rs += __shfl_xor(rs, 2, 16);
            rs += __shfl_xor(rs, 4, 16);
            rs += __shfl_xor(rs, 8, 16);
            l_r[e] = l_r[e] * alpha + rs;
            m_r[e] = mnew;
#pragma unroll
            for (int nt = 0; nt < 8; ++nt) oacc[nt][e] *= alpha;
            // stash P (bf16) for the P·V WMMA A-fragment
            bf16_t* pr = Ps + (wave * 16 + e + 8 * half) * LDPS;
            pr[ln]      = (bf16_t)e0;
            pr[16 + ln] = (bf16_t)e1;
        }

        // wave-local LDS RAW: DS ops are in-order per wave; fence + wait
        asm volatile("s_wait_dscnt 0" ::: "memory");

        v16bf pa = load_a_frag(Ps + wave * 16 * LDPS, LDPS, lane);
#pragma unroll
        for (int nt = 0; nt < 8; ++nt) {
            v16bf bv = load_bT_frag(VsT + nt * 16 * LDVT, LDVT, lane);
            oacc[nt] = wmma_bf16(pa, bv, oacc[nt]);
        }
    }

    // Epilogue: O[row][h*128 + col] = oacc / l
#pragma unroll
    for (int e = 0; e < 8; ++e) {
        float inv = 1.0f / l_r[e];
        int rowg = qb0 + wave * 16 + 8 * half + e;
#pragma unroll
        for (int nt = 0; nt < 8; ++nt) {
            O[(size_t)rowg * HV + h * VH + nt * 16 + ln] = (bf16_t)(oacc[nt][e] * inv);
        }
    }
}

// ---------------------------------------------------------------------------
// Host-side launch
// ---------------------------------------------------------------------------
static inline size_t align256(size_t x) { return (x + 255) & ~(size_t)255; }

extern "C" void kernel_launch(void* const* d_in, const int* in_sizes, int n_in,
                              void* d_out, int out_size, void* d_ws, size_t ws_size,
                              hipStream_t stream) {
    const float* h      = (const float*)d_in[0];
    const int*   pos    = (const int*)d_in[1];
    const float* Wq_a   = (const float*)d_in[2];
    const float* gq     = (const float*)d_in[3];
    const float* Wq_b   = (const float*)d_in[4];
    const float* Wkv_a  = (const float*)d_in[5];
    const float* bkv_a  = (const float*)d_in[6];
    const float* gkv    = (const float*)d_in[7];
    const float* Wkv_b  = (const float*)d_in[8];
    const float* Wo     = (const float*)d_in[9];
    float* out = (float*)d_out;

    char* ws = (char*)d_ws;
    size_t off = 0;
    auto alloc = [&](size_t bytes) -> void* {
        void* p = ws + off;
        off += align256(bytes);
        return p;
    };

    bf16_t* h_bf     = (bf16_t*)alloc((size_t)T_SEQ * D_MODEL * 2);
    bf16_t* Wq_a_bf  = (bf16_t*)alloc((size_t)D_MODEL * QR * 2);
    bf16_t* Wq_b_bf  = (bf16_t*)alloc((size_t)QR * HQK * 2);
    bf16_t* Wkv_a_bf = (bf16_t*)alloc((size_t)D_MODEL * KVL * 2);
    bf16_t* Wkv_b_bf = (bf16_t*)alloc((size_t)KVR * KVN * 2);
    bf16_t* Wo_bf    = (bf16_t*)alloc((size_t)HV * D_MODEL * 2);
    float*  qlat_f   = (float*)alloc((size_t)T_SEQ * QR * 4);
    bf16_t* qn_bf    = (bf16_t*)alloc((size_t)T_SEQ * QR * 2);
    float*  q_f      = (float*)alloc((size_t)T_SEQ * HQK * 4);
    float*  lat_f    = (float*)alloc((size_t)T_SEQ * KVL * 4);
    bf16_t* kvn_bf   = (bf16_t*)alloc((size_t)T_SEQ * KVR * 2);
    float*  kv_f     = (float*)alloc((size_t)T_SEQ * KVN * 4);
    bf16_t* Qb       = (bf16_t*)alloc((size_t)T_SEQ * H_HEADS * QK_D * 2);
    bf16_t* Kb       = (bf16_t*)alloc((size_t)T_SEQ * H_HEADS * QK_D * 2);
    bf16_t* Vb       = (bf16_t*)alloc((size_t)T_SEQ * H_HEADS * VH * 2);
    bf16_t* Ob       = (bf16_t*)alloc((size_t)T_SEQ * HV * 2);

    auto cast = [&](const float* src, bf16_t* dst, size_t n) {
        int blocks = (int)((n / 4 + 255) / 256);
        cast_f32_bf16<<<blocks, 256, 0, stream>>>(src, dst, (int)n);
    };
    cast(h,     h_bf,     (size_t)T_SEQ * D_MODEL);
    cast(Wq_a,  Wq_a_bf,  (size_t)D_MODEL * QR);
    cast(Wq_b,  Wq_b_bf,  (size_t)QR * HQK);
    cast(Wkv_a, Wkv_a_bf, (size_t)D_MODEL * KVL);
    cast(Wkv_b, Wkv_b_bf, (size_t)KVR * KVN);
    cast(Wo,    Wo_bf,    (size_t)HV * D_MODEL);

    auto gemm = [&](const bf16_t* A, const bf16_t* B, const float* bias, float* C,
                    int M, int N, int K) {
        dim3 grid((N + GTN - 1) / GTN, (M + GTM - 1) / GTM);
        gemm_bf16<<<grid, 128, 0, stream>>>(A, B, bias, C, M, N, K);
    };

    // q branch: qlat = h @ Wq_a ; qn = rmsnorm(qlat)*gq ; q = qn @ Wq_b
    gemm(h_bf, Wq_a_bf, nullptr, qlat_f, T_SEQ, QR, D_MODEL);
    rmsnorm_to_bf16<<<T_SEQ, 256, 0, stream>>>(qlat_f, QR, gq, qn_bf, QR);
    gemm(qn_bf, Wq_b_bf, nullptr, q_f, T_SEQ, HQK, QR);

    // kv branch: lat = h @ Wkv_a + b ; kvn = rmsnorm(lat[:,:512])*gkv ; kv = kvn @ Wkv_b
    gemm(h_bf, Wkv_a_bf, bkv_a, lat_f, T_SEQ, KVL, D_MODEL);
    rmsnorm_to_bf16<<<T_SEQ, 256, 0, stream>>>(lat_f, KVL, gkv, kvn_bf, KVR);
    gemm(kvn_bf, Wkv_b_bf, nullptr, kv_f, T_SEQ, KVN, KVR);

    // assemble Q/K/V with RoPE
    {
        dim3 grid(T_SEQ, H_HEADS);
        assemble_qkv<<<grid, 160, 0, stream>>>(q_f, kv_f, lat_f, pos, Qb, Kb, Vb);
    }

    // causal flash attention
    {
        dim3 grid(T_SEQ / FA_QB, H_HEADS);
        flash_attn<<<grid, 128, 0, stream>>>(Qb, Kb, Vb, Ob);
    }

    // output projection
    gemm(Ob, Wo_bf, nullptr, out, T_SEQ, D_MODEL, HV);
}
